// QCNet_65481071402916
// MI455X (gfx1250) — matrix operand order repaired and checked by reading, not compiled, not run on hardware
//
#include <hip/hip_runtime.h>
#include <hip/hip_bf16.h>

typedef __attribute__((ext_vector_type(16))) _Float16 v16h;
typedef __attribute__((ext_vector_type(8)))  _Float16 v8h;
typedef __attribute__((ext_vector_type(8)))  float    v8f;

#define BN_EPS 1e-5f

// ---------------------------------------------------------------------------
// Kernel 1: pack FC weight [N,K] (f32, row-major) into WMMA B-fragment layout:
// frag[ntile][kchunk][lane][j] (f16), j = 0..15.
//   n = ntile*16 + (lane & 15)
//   k = kchunk*32 + (lane >= 16 ? 8 : 0) + (j < 8 ? j : 16 + (j - 8))      (matches 16-bit B 32x16 layout)
// Zero-padded outside [N,K].
// ---------------------------------------------------------------------------
__global__ void pack_w_kernel(const float* __restrict__ w, int N, int K,
                              int kchunks, int total, _Float16* __restrict__ frag) {
    int idx = blockIdx.x * blockDim.x + threadIdx.x;
    if (idx >= total) return;
    int j     = idx & 15;
    int lane  = (idx >> 4) & 31;
    int kc    = (idx >> 9) % kchunks;
    int ntile = idx / (kchunks * 512);
    int n = ntile * 16 + (lane & 15);
    int k = kc * 32 + ((lane >> 4) ? 8 : 0) + (j < 8 ? j : 16 + (j - 8));
    float v = (n < N && k < K) ? w[(size_t)n * K + k] : 0.0f;
    frag[idx] = (_Float16)v;
}

// ---------------------------------------------------------------------------
// Kernel 2: fused conv1+BN+ReLU+pool + conv2+BN+ReLU+pool, one image per block.
// Input x[img] streamed into LDS once; output flattened [540] -> f16 [544].
// ---------------------------------------------------------------------------
__global__ void __launch_bounds__(256)
conv_fused_kernel(const float* __restrict__ x,
                  const float* __restrict__ cw1, const float* __restrict__ cb1,
                  const float* __restrict__ g1,  const float* __restrict__ bb1,
                  const float* __restrict__ m1,  const float* __restrict__ v1,
                  const float* __restrict__ cw2, const float* __restrict__ cb2,
                  const float* __restrict__ g2,  const float* __restrict__ bb2,
                  const float* __restrict__ m2,  const float* __restrict__ v2,
                  _Float16* __restrict__ act) {
    __shared__ float sIn[3 * 32 * 32];   // 3072
    __shared__ float sW1[6 * 3 * 25];    // 450
    __shared__ float sS1[6], sO1[6];
    __shared__ float sC1[6 * 15 * 15];   // 1350
    __shared__ float sP1[6 * 14 * 14];   // 1176
    __shared__ float sW2[15 * 6 * 9];    // 810
    __shared__ float sS2[15], sO2[15];
    __shared__ float sC2[15 * 7 * 7];    // 735

    const int img = blockIdx.x;
    const int t   = threadIdx.x;

    const float* xin = x + (size_t)img * 3072;
    for (int i = t; i < 3072; i += 256) sIn[i] = xin[i];
    for (int i = t; i < 450;  i += 256) sW1[i] = cw1[i];
    for (int i = t; i < 810;  i += 256) sW2[i] = cw2[i];
    if (t < 6) {
        float s = g1[t] * __frsqrt_rn(v1[t] + BN_EPS);
        sS1[t] = s;
        sO1[t] = bb1[t] + (cb1[t] - m1[t]) * s;   // y = conv*s + O
    }
    if (t < 15) {
        float s = g2[t] * __frsqrt_rn(v2[t] + BN_EPS);
        sS2[t] = s;
        sO2[t] = bb2[t] + (cb2[t] - m2[t]) * s;
    }
    __syncthreads();

    // conv1: 6 x 15 x 15, 5x5 s2 p1
    for (int o = t; o < 1350; o += 256) {
        int ch = o / 225, r = o % 225, oy = r / 15, ox = r % 15;
        int iy0 = oy * 2 - 1, ix0 = ox * 2 - 1;
        float acc = 0.0f;
        const float* w = &sW1[ch * 75];
        #pragma unroll
        for (int ci = 0; ci < 3; ci++) {
            const float* ip = &sIn[ci * 1024];
            #pragma unroll
            for (int ky = 0; ky < 5; ky++) {
                int iy = iy0 + ky;
                if ((unsigned)iy >= 32u) continue;
                #pragma unroll
                for (int kx = 0; kx < 5; kx++) {
                    int ix = ix0 + kx;
                    if ((unsigned)ix >= 32u) continue;
                    acc = fmaf(ip[iy * 32 + ix], w[ci * 25 + ky * 5 + kx], acc);
                }
            }
        }
        float y = fmaf(acc, sS1[ch], sO1[ch]);
        sC1[o] = y > 0.0f ? y : 0.0f;
    }
    __syncthreads();

    // maxpool 2x2 s1: 6 x 14 x 14
    for (int o = t; o < 1176; o += 256) {
        int ch = o / 196, r = o % 196, y = r / 14, xp = r % 14;
        const float* p = &sC1[ch * 225 + y * 15 + xp];
        sP1[o] = fmaxf(fmaxf(p[0], p[1]), fmaxf(p[15], p[16]));
    }
    __syncthreads();

    // conv2: 15 x 7 x 7, 3x3 s2 p1
    for (int o = t; o < 735; o += 256) {
        int ch = o / 49, r = o % 49, oy = r / 7, ox = r % 7;
        int iy0 = oy * 2 - 1, ix0 = ox * 2 - 1;
        float acc = 0.0f;
        const float* w = &sW2[ch * 54];
        #pragma unroll
        for (int ci = 0; ci < 6; ci++) {
            const float* ip = &sP1[ci * 196];
            #pragma unroll
            for (int ky = 0; ky < 3; ky++) {
                int iy = iy0 + ky;
                if ((unsigned)iy >= 14u) continue;
                #pragma unroll
                for (int kx = 0; kx < 3; kx++) {
                    int ix = ix0 + kx;
                    if ((unsigned)ix >= 14u) continue;
                    acc = fmaf(ip[iy * 14 + ix], w[ci * 9 + ky * 3 + kx], acc);
                }
            }
        }
        float y = fmaf(acc, sS2[ch], sO2[ch]);
        sC2[o] = y > 0.0f ? y : 0.0f;
    }
    __syncthreads();

    // maxpool 2x2 s1 -> flatten [15*6*6=540] -> f16, pad to 544
    _Float16* arow = act + (size_t)img * 544;
    for (int o = t; o < 544; o += 256) {
        if (o < 540) {
            int ch = o / 36, r = o % 36, y = r / 6, xp = r % 6;
            const float* p = &sC2[ch * 49 + y * 7 + xp];
            arow[o] = (_Float16)fmaxf(fmaxf(p[0], p[1]), fmaxf(p[7], p[8]));
        } else {
            arow[o] = (_Float16)0.0f;
        }
    }
}

// ---------------------------------------------------------------------------
// Kernel 3: WMMA GEMM + bias + ReLU.
//   A:    f16 [B, lda] (row-major, K padded to kchunks*32, pad = 0)
//   Wfrag: pre-swizzled B-fragments [ntiles][kchunks][32 lanes][16 f16]
//   Out:  f16 [B, ldo]
// grid.x = B/16 (M tile); blockDim = (32, ntiles): one wave per N tile.
// A-fragment per ISA 16-bit 16x32 layout:
//   lane<16: elems 0..7 = K 0..7,  elems 8..15 = K 16..23
//   lane>=16: elems 0..7 = K 8..15, elems 8..15 = K 24..31
// ---------------------------------------------------------------------------
__global__ void wmma_fc_kernel(const _Float16* __restrict__ A, int lda,
                               const _Float16* __restrict__ Wfrag,
                               const float* __restrict__ bias, int nbias,
                               _Float16* __restrict__ Out, int ldo, int kchunks) {
    const int lane  = threadIdx.x;          // 0..31
    const int ntile = threadIdx.y;
    const int mtile = blockIdx.x;
    const int hiSel = (lane >> 4) & 1;

    const _Float16* Arow = A + ((size_t)mtile * 16 + (lane & 15)) * lda;
    const _Float16* Wp0  = Wfrag + ((size_t)ntile * kchunks) * 512 + (size_t)lane * 16;

    v8f c = {};
    for (int kc = 0; kc < kchunks; kc++) {
        const int kbase = kc * 32;
        const int klo = kbase + (hiSel ? 8 : 0);
        const int khi = kbase + (hiSel ? 24 : 16);
        __builtin_prefetch(Arow + kbase + 64, 0, 1);   // global_prefetch_b8
        v8h alo = *(const v8h*)(Arow + klo);
        v8h ahi = *(const v8h*)(Arow + khi);
        v16h a = __builtin_shufflevector(alo, ahi, 0, 1, 2, 3, 4, 5, 6, 7,
                                                   8, 9, 10, 11, 12, 13, 14, 15);
        const _Float16* wp = Wp0 + (size_t)kc * 512;
        v8h blo = *(const v8h*)(wp);
        v8h bhi = *(const v8h*)(wp + 8);
        v16h b = __builtin_shufflevector(blo, bhi, 0, 1, 2, 3, 4, 5, 6, 7,
                                                   8, 9, 10, 11, 12, 13, 14, 15);
        // D = A x B + C  -> v_wmma_f32_16x16x32_f16
        c = __builtin_amdgcn_wmma_f32_16x16x32_f16(false, a, false, b,
                                                   (short)0, c, false, false);
    }

    // Epilogue per 16x16 f32 C/D layout: VGPR r -> M = r + (lane>=16 ? 8 : 0), N = lane&15
    const int n  = ntile * 16 + (lane & 15);
    const float bv = (n < nbias) ? bias[n] : 0.0f;
    #pragma unroll
    for (int r = 0; r < 8; r++) {
        int m = mtile * 16 + r + (hiSel ? 8 : 0);
        float y = c[r] + bv;
        y = y > 0.0f ? y : 0.0f;
        Out[(size_t)m * ldo + n] = (_Float16)y;
    }
}

// ---------------------------------------------------------------------------
// Kernel 4: fc3 + residual head + quantum expval + sigmoid -> [B,2] f32.
// ---------------------------------------------------------------------------
__global__ void fc_final_kernel(const _Float16* __restrict__ h1,
                                const _Float16* __restrict__ h2,
                                const float* __restrict__ w3, const float* __restrict__ b3,
                                const float* __restrict__ wr, const float* __restrict__ br,
                                float* __restrict__ out, int B) {
    int i = blockIdx.x * blockDim.x + threadIdx.x;
    if (i >= B) return;
    const _Float16* r1 = h1 + (size_t)i * 128;
    const _Float16* r2 = h2 + (size_t)i * 96;
    float acc = b3[0] + br[0];
    for (int k = 0; k < 84; k++)  acc = fmaf((float)r2[k], w3[k], acc);
    for (int k = 0; k < 120; k++) acc = fmaf((float)r1[k], wr[k], acc);

    // Quantum circuit: H(0); CNOT(0,1); RX(acc,0); RZ(pi/2,1); <Z_0>.
    // |a00|^2+|a01|^2 = (c^2+s^2)/2, |a10|^2+|a11|^2 = (s^2+c^2)/2 -> q = 0.
    float hlf = acc * 0.5f;
    float cs = __cosf(hlf), sn = __sinf(hlf);
    float p0 = 0.5f * (cs * cs) + 0.5f * (sn * sn);
    float p1 = 0.5f * (sn * sn) + 0.5f * (cs * cs);
    float q = p0 - p1;
    float prob = 1.0f / (1.0f + __expf(-q));
    out[2 * (size_t)i + 0] = prob;
    out[2 * (size_t)i + 1] = 1.0f - prob;
}

// ---------------------------------------------------------------------------
extern "C" void kernel_launch(void* const* d_in, const int* in_sizes, int n_in,
                              void* d_out, int out_size, void* d_ws, size_t ws_size,
                              hipStream_t stream) {
    const float* x      = (const float*)d_in[0];
    const float* cw1    = (const float*)d_in[1];
    const float* cb1    = (const float*)d_in[2];
    const float* g1     = (const float*)d_in[3];
    const float* bb1    = (const float*)d_in[4];
    const float* m1     = (const float*)d_in[5];
    const float* v1     = (const float*)d_in[6];
    const float* cw2    = (const float*)d_in[7];
    const float* cb2    = (const float*)d_in[8];
    const float* g2     = (const float*)d_in[9];
    const float* bb2    = (const float*)d_in[10];
    const float* m2     = (const float*)d_in[11];
    const float* v2     = (const float*)d_in[12];
    const float* fc1_w  = (const float*)d_in[13];
    const float* fc1_b  = (const float*)d_in[14];
    const float* fc2_w  = (const float*)d_in[15];
    const float* fc2_b  = (const float*)d_in[16];
    const float* fc3_w  = (const float*)d_in[17];
    const float* fc3_b  = (const float*)d_in[18];
    const float* res_w  = (const float*)d_in[19];
    const float* res_b  = (const float*)d_in[20];
    float* out = (float*)d_out;

    const int B = in_sizes[0] / (3 * 32 * 32);   // 16384

    // FC1: K=540 -> 17 chunks of 32 (pad 544); N=120 -> 8 tiles (pad 128)
    const int KC1 = 17, NT1 = 8;
    // FC2: K=120 -> 4 chunks (pad 128); N=84 -> 6 tiles (pad 96)
    const int KC2 = 4, NT2 = 6;

    auto align256 = [](size_t v) { return (v + 255) & ~(size_t)255; };
    char* base = (char*)d_ws;
    size_t off = 0;
    _Float16* act = (_Float16*)(base + off); off = align256(off + (size_t)B * 544 * sizeof(_Float16));
    _Float16* h1  = (_Float16*)(base + off); off = align256(off + (size_t)B * 128 * sizeof(_Float16));
    _Float16* h2  = (_Float16*)(base + off); off = align256(off + (size_t)B * 96  * sizeof(_Float16));
    _Float16* w1f = (_Float16*)(base + off); off = align256(off + (size_t)NT1 * KC1 * 512 * sizeof(_Float16));
    _Float16* w2f = (_Float16*)(base + off); off = align256(off + (size_t)NT2 * KC2 * 512 * sizeof(_Float16));
    (void)ws_size; (void)n_in; (void)out_size;

    // Pack FC weights into WMMA B-fragment layout.
    {
        int total1 = NT1 * KC1 * 512;
        pack_w_kernel<<<(total1 + 255) / 256, 256, 0, stream>>>(fc1_w, 120, 540, KC1, total1, w1f);
        int total2 = NT2 * KC2 * 512;
        pack_w_kernel<<<(total2 + 255) / 256, 256, 0, stream>>>(fc2_w, 84, 120, KC2, total2, w2f);
    }

    // Fused conv stack: one image per 256-thread block.
    conv_fused_kernel<<<B, 256, 0, stream>>>(x, cw1, cb1, g1, bb1, m1, v1,
                                             cw2, cb2, g2, bb2, m2, v2, act);

    // fc1: [B,544] x [544,128] -> h1 [B,128] (f16), bias+ReLU
    wmma_fc_kernel<<<dim3(B / 16), dim3(32, NT1), 0, stream>>>(
        act, 544, w1f, fc1_b, 120, h1, 128, KC1);

    // fc2: [B,128] x [128,96] -> h2 [B,96] (f16), bias+ReLU
    wmma_fc_kernel<<<dim3(B / 16), dim3(32, NT2), 0, stream>>>(
        h1, 128, w2f, fc2_b, 84, h2, 96, KC2);

    // Heads + quantum + sigmoid -> [B,2]
    fc_final_kernel<<<(B + 255) / 256, 256, 0, stream>>>(
        h1, h2, fc3_w, fc3_b, res_w, res_b, out, B);
}
